// SwinBackbone_13099650252911
// MI455X (gfx1250) — compile-verified, matrix operand order checked by reading, not run on hardware
//
#include <hip/hip_runtime.h>
#include <cstdint>
#include <cmath>

typedef __attribute__((ext_vector_type(16))) __bf16 v16bf;
typedef __attribute__((ext_vector_type(8)))  float  v8f;
typedef unsigned short u16;

__device__ __forceinline__ __bf16 us2bf(u16 u) { return __builtin_bit_cast(__bf16, u); }
__device__ __forceinline__ u16    f2us(float f){ return __builtin_bit_cast(u16, (__bf16)f); }

// region id along one axis for the shifted-window mask (replicates the
// reference's slice construction; later slices overwrite earlier ones).
__device__ __forceinline__ int regionOf(int a, int L, int w, int s) {
  if (s == 0) return 0;
  if (a < L - w) return 0;
  if (a < L - s) return 1;
  return 2;
}

// ---------------------------------------------------------------- cvt f32->bf16
__global__ void k_cvt(const float* __restrict__ in, u16* __restrict__ out, int n) {
  int i = blockIdx.x * blockDim.x + threadIdx.x;
  if (i < n) out[i] = f2us(in[i]);
}

// ---------------------------------------------------------------- patch im2col
// x: (1,16,224,224,3) f32 -> out [25088, 96] bf16, col = ((t*4+p)*4+q)*3+c
__global__ void k_patchify(const float* __restrict__ x, u16* __restrict__ out) {
  int gid = blockIdx.x * blockDim.x + threadIdx.x;
  const int total = 25088 * 96;
  if (gid >= total) return;
  int row = gid / 96, col = gid - row * 96;
  int c = col % 3, q = (col / 3) & 3, p = (col / 12) & 3, t = col / 48;
  int w = row % 56, h = (row / 56) % 56, d = row / (56 * 56);
  size_t src = ((((size_t)(d * 2 + t) * 224) + (h * 4 + p)) * 224 + (w * 4 + q)) * 3 + c;
  out[gid] = f2us(x[src]);
}

// ---------------------------------------------------------------- patch merge gather
// out [D*(H/2)*(W/2), 4C] f32; quad order: (he,we),(ho,we),(he,wo),(ho,wo)
__global__ void k_merge_gather(const float* __restrict__ x, float* __restrict__ out,
                               int D, int H, int W, int C) {
  int H2 = H / 2, W2 = W / 2;
  size_t total = (size_t)D * H2 * W2 * 4 * C;
  size_t gid = (size_t)blockIdx.x * blockDim.x + threadIdx.x;
  if (gid >= total) return;
  int C4 = 4 * C;
  int row = (int)(gid / C4), col = (int)(gid - (size_t)row * C4);
  int quad = col / C, c = col - quad * C;
  int w2 = row % W2, h2 = (row / W2) % H2, d = row / (W2 * H2);
  int hh = 2 * h2 + (quad & 1);
  int ww = 2 * w2 + (quad >> 1);
  out[gid] = x[(((size_t)d * H + hh) * W + ww) * C + c];
}

// ---------------------------------------------------------------- LayerNorm (+ optional window-gather)
// One wave per row. If useMap: output row index is in window order (with roll),
// source row is the corresponding original token.
__global__ void k_ln_gather(const float* __restrict__ x, const float* __restrict__ g,
                            const float* __restrict__ bta, float* __restrict__ outF,
                            u16* __restrict__ outB, int rows, int C,
                            int D, int H, int W, int wd, int wh, int ww,
                            int sd, int sh, int sw, int useMap) {
  int row  = blockIdx.x * (blockDim.x >> 5) + (threadIdx.x >> 5);
  int lane = threadIdx.x & 31;
  if (row >= rows) return;
  int src = row;
  if (useMap) {
    int N = wd * wh * ww;
    int w = row / N, t = row - w * N;
    int nH = H / wh, nWw = W / ww;
    int bW = w % nWw, bH = (w / nWw) % nH, bD = w / (nWw * nH);
    int iw = t % ww, ih = (t / ww) % wh, id = t / (ww * wh);
    int ad = (bD * wd + id + sd) % D;
    int ah = (bH * wh + ih + sh) % H;
    int aw = (bW * ww + iw + sw) % W;
    src = (ad * H + ah) * W + aw;
  }
  const float* xr = x + (size_t)src * C;
  float s = 0.f;
  for (int i = lane; i < C; i += 32) s += xr[i];
#pragma unroll
  for (int o = 16; o > 0; o >>= 1) s += __shfl_xor(s, o, 32);
  float mean = s / (float)C;
  float var = 0.f;
  for (int i = lane; i < C; i += 32) { float d = xr[i] - mean; var += d * d; }
#pragma unroll
  for (int o = 16; o > 0; o >>= 1) var += __shfl_xor(var, o, 32);
  float rstd = rsqrtf(var / (float)C + 1e-5f);
  for (int i = lane; i < C; i += 32) {
    float v = (xr[i] - mean) * rstd * g[i] + bta[i];
    if (outF) outF[(size_t)row * C + i] = v;
    if (outB) outB[(size_t)row * C + i] = f2us(v);
  }
}

// ---------------------------------------------------------------- scatter-add (window un-partition + roll back)
__global__ void k_scatter_add(float* __restrict__ x, const float* __restrict__ y,
                              int T, int C, int D, int H, int W,
                              int wd, int wh, int ww, int sd, int sh, int sw) {
  size_t gid = (size_t)blockIdx.x * blockDim.x + threadIdx.x;
  if (gid >= (size_t)T * C) return;
  int t = (int)(gid / C), c = (int)(gid - (size_t)t * C);
  int N = wd * wh * ww;
  int w = t / N, u = t - w * N;
  int nH = H / wh, nWw = W / ww;
  int bW = w % nWw, bH = (w / nWw) % nH, bD = w / (nWw * nH);
  int iw = u % ww, ih = (u / ww) % wh, id = u / (ww * wh);
  int ad = (bD * wd + id + sd) % D;
  int ah = (bH * wh + ih + sh) % H;
  int aw = (bW * ww + iw + sw) % W;
  int src = (ad * H + ah) * W + aw;
  x[(size_t)src * C + c] += y[gid];
}

// ---------------------------------------------------------------- GEMM: C = A(bf16)[M,K] @ B(bf16)[K,N] + bias
// 8 waves, each computes a 32x32 register tile (2x2 WMMAs) -> block tile 128(M) x 64(N).
// act: 0 none, 1 exact GELU. outF (f32, optional, addTo accumulates), outB (bf16, optional).
__global__ __launch_bounds__(256) void k_gemm_bf16(
    const u16* __restrict__ A, const u16* __restrict__ B, const float* __restrict__ bias,
    float* __restrict__ outF, u16* __restrict__ outB,
    int M, int N, int K, int act, int addTo) {
  __shared__ u16 As[128][36];  // A tile (128 rows x 32 k), padded
  __shared__ u16 Bs[64][36];   // B tile transposed: Bs[n][k]
  const int tid = threadIdx.x, lane = tid & 31, wid = tid >> 5;
  const int wm = wid & 3, wn = wid >> 2;            // 4 (M) x 2 (N) waves
  const int bm = blockIdx.y * 128, bn = blockIdx.x * 64;
  const int l15 = lane & 15, half = lane >> 4, kb = half * 8;
  v8f acc[2][2];
#pragma unroll
  for (int mi = 0; mi < 2; mi++)
#pragma unroll
    for (int ni = 0; ni < 2; ni++) {
      v8f z = {0.f, 0.f, 0.f, 0.f, 0.f, 0.f, 0.f, 0.f};
      acc[mi][ni] = z;
    }
  for (int k0 = 0; k0 < K; k0 += 32) {
    // speculative prefetch of the next K tile (global_prefetch_b8 path)
    if (k0 + 32 < K) {
      int pr = bm + (tid >> 1); if (pr >= M) pr = M - 1;
      __builtin_prefetch(A + (size_t)pr * K + (k0 + 32) + (tid & 1) * 16, 0, 1);
      int pk = k0 + 32 + (tid >> 3); if (pk >= K) pk = K - 1;
      int pn = bn + (tid & 7) * 8; if (pn >= N) pn = N - 1;
      __builtin_prefetch(B + (size_t)pk * N + pn, 0, 1);
    }
    for (int i = tid; i < 128 * 32; i += 256) {
      int r = i >> 5, cc = i & 31, gr = bm + r, gk = k0 + cc;
      As[r][cc] = (gr < M && gk < K) ? A[(size_t)gr * K + gk] : (u16)0;
    }
    for (int i = tid; i < 64 * 32; i += 256) {
      int n = i >> 5, kk = i & 31, gn = bn + n, gk = k0 + kk;
      Bs[n][kk] = (gn < N && gk < K) ? B[(size_t)gk * N + gn] : (u16)0;
    }
    __syncthreads();
    v16bf a[2], b[2];
#pragma unroll
    for (int mi = 0; mi < 2; mi++) {
      int ar = wm * 32 + mi * 16 + l15;
#pragma unroll
      for (int i = 0; i < 8; i++) {
        a[mi][i]     = us2bf(As[ar][kb + i]);
        a[mi][8 + i] = us2bf(As[ar][kb + 16 + i]);
      }
    }
#pragma unroll
    for (int ni = 0; ni < 2; ni++) {
      int bc = wn * 32 + ni * 16 + l15;
#pragma unroll
      for (int i = 0; i < 8; i++) {
        b[ni][i]     = us2bf(Bs[bc][kb + i]);
        b[ni][8 + i] = us2bf(Bs[bc][kb + 16 + i]);
      }
    }
#pragma unroll
    for (int mi = 0; mi < 2; mi++)
#pragma unroll
      for (int ni = 0; ni < 2; ni++)
        acc[mi][ni] = __builtin_amdgcn_wmma_f32_16x16x32_bf16(
            false, a[mi], false, b[ni], (short)0, acc[mi][ni], false, false);
    __syncthreads();
  }
#pragma unroll
  for (int ni = 0; ni < 2; ni++) {
    int gcol = bn + wn * 32 + ni * 16 + l15;
    if (gcol < N) {
      float bv = bias ? bias[gcol] : 0.f;
#pragma unroll
      for (int mi = 0; mi < 2; mi++) {
#pragma unroll
        for (int r = 0; r < 8; r++) {
          int grow = bm + wm * 32 + mi * 16 + r + half * 8;
          if (grow < M) {
            float v = acc[mi][ni][r] + bv;
            if (act == 1) v = 0.5f * v * (1.f + erff(v * 0.70710678118654752f));
            size_t o = (size_t)grow * N + gcol;
            if (outF) outF[o] = addTo ? (outF[o] + v) : v;
            if (outB) outB[o] = f2us(v);
          }
        }
      }
    }
  }
}

// ---------------------------------------------------------------- fused window attention
// One wave per (window, head, 16-query tile). qkv: [nWin*N, 3C] bf16 window-ordered.
// out: [nWin*N, C] bf16 window-ordered. Scores in LDS; softmax + rel-bias + shift mask fused.
__global__ __launch_bounds__(32) void k_attn(
    const u16* __restrict__ qkv, const float* __restrict__ rpb, u16* __restrict__ out,
    int nWin, int nh, int N, int C, float scale,
    int D, int H, int W, int wd, int wh, int ww, int sd, int sh, int sw) {
  __shared__ float sc[16][400];   // scores, 16 queries x up to 400 keys
  __shared__ u16   pb[16][416];   // probabilities bf16, padded to 32-key tiles
  const int lane = threadIdx.x, l15 = lane & 15, half = lane >> 4, kb = half * 8;
  const int QT = (N + 15) / 16;
  int bid = blockIdx.x;
  int qt = bid % QT;
  int t2 = bid / QT;
  int head = t2 % nh;
  int w = t2 / nh;
  const int C3 = 3 * C, h32 = head * 32;

  for (int i = lane; i < 16 * 416; i += 32) ((u16*)pb)[i] = (u16)0;
  __syncthreads();

  // Q fragment (A-matrix 16x32)
  v16bf aq;
  {
    int qrow = qt * 16 + l15;
    size_t base = ((size_t)(w * N + (qrow < N ? qrow : 0))) * C3 + h32;
#pragma unroll
    for (int i = 0; i < 8; i++) {
      aq[i]     = (qrow < N) ? us2bf(qkv[base + kb + i])      : (__bf16)0.f;
      aq[8 + i] = (qrow < N) ? us2bf(qkv[base + kb + 16 + i]) : (__bf16)0.f;
    }
  }
  const int nH = H / wh, nWw = W / ww;
  const int bW = w % nWw, bH = (w / nWw) % nH, bD = w / (nWw * nH);
  const bool masked = (sd | sh | sw) != 0;
  const int rh2 = 2 * wh - 1, rw2 = 2 * ww - 1;

  // ---- scores: Q @ K^T, one 16x16 tile per wmma
  for (int kt = 0; kt < QT; kt++) {
    v16bf bk;
    int key = kt * 16 + l15;
    size_t kbase = ((size_t)(w * N + (key < N ? key : 0))) * C3 + C + h32;
#pragma unroll
    for (int i = 0; i < 8; i++) {
      bk[i]     = (key < N) ? us2bf(qkv[kbase + kb + i])      : (__bf16)0.f;
      bk[8 + i] = (key < N) ? us2bf(qkv[kbase + kb + 16 + i]) : (__bf16)0.f;
    }
    v8f s = {0.f, 0.f, 0.f, 0.f, 0.f, 0.f, 0.f, 0.f};
    s = __builtin_amdgcn_wmma_f32_16x16x32_bf16(false, aq, false, bk, (short)0, s, false, false);
    int col = kt * 16 + l15;
    if (col < N) {
      int kiw = col % ww, kih = (col / ww) % wh, kid = col / (ww * wh);
      int rKd = regionOf(bD * wd + kid, D, wd, sd);
      int rKh = regionOf(bH * wh + kih, H, wh, sh);
      int rKw = regionOf(bW * ww + kiw, W, ww, sw);
#pragma unroll
      for (int r = 0; r < 8; r++) {
        int m = r + half * 8;
        int qi = qt * 16 + m;
        if (qi < N) {
          int qiw = qi % ww, qih = (qi / ww) % wh, qid = qi / (ww * wh);
          int rpi = ((qid - kid + wd - 1) * rh2 + (qih - kih + wh - 1)) * rw2 + (qiw - kiw + ww - 1);
          float v = s[r] * scale + rpb[(size_t)rpi * nh + head];
          if (masked) {
            int rQd = regionOf(bD * wd + qid, D, wd, sd);
            int rQh = regionOf(bH * wh + qih, H, wh, sh);
            int rQw = regionOf(bW * ww + qiw, W, ww, sw);
            if (rQd != rKd || rQh != rKh || rQw != rKw) v += -100.f;
          }
          sc[m][col] = v;
        }
      }
    }
  }
  __syncthreads();

  // ---- softmax per query row (wave-cooperative)
  for (int m = 0; m < 16; m++) {
    if (qt * 16 + m >= N) break;
    float mx = -3.0e38f;
    for (int i = lane; i < N; i += 32) mx = fmaxf(mx, sc[m][i]);
#pragma unroll
    for (int o = 16; o > 0; o >>= 1) mx = fmaxf(mx, __shfl_xor(mx, o, 32));
    float sum = 0.f;
    for (int i = lane; i < N; i += 32) { float e = __expf(sc[m][i] - mx); sc[m][i] = e; sum += e; }
#pragma unroll
    for (int o = 16; o > 0; o >>= 1) sum += __shfl_xor(sum, o, 32);
    float inv = 1.f / sum;
    for (int i = lane; i < N; i += 32) pb[m][i] = f2us(sc[m][i] * inv);
  }
  __syncthreads();

  // ---- PV: probs(16 x 32keys) @ V(32keys x 32chan) as two 16x16 wmmas per tile
  v8f c0 = {0.f, 0.f, 0.f, 0.f, 0.f, 0.f, 0.f, 0.f};
  v8f c1 = {0.f, 0.f, 0.f, 0.f, 0.f, 0.f, 0.f, 0.f};
  int PT = (N + 31) / 32;
  for (int kt2 = 0; kt2 < PT; kt2++) {
    v16bf ap, bv0, bv1;
#pragma unroll
    for (int i = 0; i < 8; i++) {
      ap[i]     = us2bf(pb[l15][kt2 * 32 + kb + i]);
      ap[8 + i] = us2bf(pb[l15][kt2 * 32 + kb + 16 + i]);
    }
#pragma unroll
    for (int i = 0; i < 8; i++) {
      int key0 = kt2 * 32 + kb + i, key1 = key0 + 16;
      size_t b0 = ((size_t)(w * N + (key0 < N ? key0 : 0))) * C3 + 2 * C + h32;
      size_t b1 = ((size_t)(w * N + (key1 < N ? key1 : 0))) * C3 + 2 * C + h32;
      bv0[i]     = (key0 < N) ? us2bf(qkv[b0 + l15])      : (__bf16)0.f;
      bv0[8 + i] = (key1 < N) ? us2bf(qkv[b1 + l15])      : (__bf16)0.f;
      bv1[i]     = (key0 < N) ? us2bf(qkv[b0 + 16 + l15]) : (__bf16)0.f;
      bv1[8 + i] = (key1 < N) ? us2bf(qkv[b1 + 16 + l15]) : (__bf16)0.f;
    }
    c0 = __builtin_amdgcn_wmma_f32_16x16x32_bf16(false, ap, false, bv0, (short)0, c0, false, false);
    c1 = __builtin_amdgcn_wmma_f32_16x16x32_bf16(false, ap, false, bv1, (short)0, c1, false, false);
  }
#pragma unroll
  for (int r = 0; r < 8; r++) {
    int qi = qt * 16 + r + half * 8;
    if (qi < N) {
      size_t o = ((size_t)(w * N + qi)) * C + h32;
      out[o + l15]      = f2us(c0[r]);
      out[o + 16 + l15] = f2us(c1[r]);
    }
  }
}

// ================================================================ host orchestration
extern "C" void kernel_launch(void* const* d_in, const int* in_sizes, int n_in,
                              void* d_out, int out_size, void* d_ws, size_t ws_size,
                              hipStream_t stream) {
  (void)in_sizes; (void)n_in; (void)out_size; (void)ws_size;
  const float* x_in = (const float*)d_in[0];
  const float* pe_w = (const float*)d_in[1];
  const float* pe_b = (const float*)d_in[2];
  const float* pn_g = (const float*)d_in[3];
  const float* pn_b = (const float*)d_in[4];

  // workspace arenas (stream-sequential reuse)
  char* ws = (char*)d_ws;
  float* xbuf = (float*)ws;                                        // residual stream, <= 25088*96 f32
  float* ybuf = (float*)(ws + (size_t)2408448 * 4);                // <= 25088*384 f32
  u16*   y16  = (u16*)(ws + (size_t)2408448 * 4 + (size_t)9633792 * 4); // <= 25088*384 bf16
  u16*   abuf = y16 + 9633792;                                     // <= 2408448 bf16 (LN / gather out)
  u16*   wbuf = abuf + 2408448;                                    // <= 768*3072 bf16 (weight scratch)
  u16*   aout = wbuf + 2359296;                                    // <= 25088*96 bf16 (attention out)

  // ---- patch embed + patch norm
  {
    int total = 25088 * 96;
    k_patchify<<<(total + 255) / 256, 256, 0, stream>>>(x_in, abuf);
    k_cvt<<<(96 * 96 + 255) / 256, 256, 0, stream>>>(pe_w, wbuf, 96 * 96);
    dim3 g((96 + 63) / 64, (25088 + 127) / 128);
    k_gemm_bf16<<<g, 256, 0, stream>>>(abuf, wbuf, pe_b, xbuf, (u16*)nullptr, 25088, 96, 96, 0, 0);
    k_ln_gather<<<(25088 + 7) / 8, 256, 0, stream>>>(xbuf, pn_g, pn_b, xbuf, (u16*)nullptr,
                                                     25088, 96, 0, 0, 0, 0, 0, 0, 0, 0, 0, 0);
  }

  const int DEP[4] = {2, 2, 6, 2};
  const int NH[4]  = {3, 6, 12, 24};
  int D = 8, H = 56, W = 56, C = 96;
  int p = 5;
  for (int i = 0; i < 4; i++) {
    for (int j = 0; j < DEP[i]; j++) {
      const float* n1g   = (const float*)d_in[p + 0];
      const float* n1b   = (const float*)d_in[p + 1];
      const float* qkv_w = (const float*)d_in[p + 2];
      const float* qkv_b = (const float*)d_in[p + 3];
      const float* rpb   = (const float*)d_in[p + 4];
      const float* prj_w = (const float*)d_in[p + 5];
      const float* prj_b = (const float*)d_in[p + 6];
      const float* n2g   = (const float*)d_in[p + 7];
      const float* n2b   = (const float*)d_in[p + 8];
      const float* fc1_w = (const float*)d_in[p + 9];
      const float* fc1_b = (const float*)d_in[p + 10];
      const float* fc2_w = (const float*)d_in[p + 11];
      const float* fc2_b = (const float*)d_in[p + 12];
      p += 13;
      bool shifted = (j & 1) != 0;
      int wd = (D <= 8) ? D : 8, wh = (H <= 7) ? H : 7, wwv = (W <= 7) ? W : 7;
      int sd = (shifted && D > 8) ? 4 : 0;
      int sh = (shifted && H > 7) ? 3 : 0;
      int sw = (shifted && W > 7) ? 3 : 0;
      int T = D * H * W, N = wd * wh * wwv, nWin = T / N, nh = NH[i];
      float scale = 1.0f / sqrtf((float)(C / nh));

      // LN1 fused with window partition (+roll) -> bf16
      k_ln_gather<<<(T + 7) / 8, 256, 0, stream>>>(xbuf, n1g, n1b, (float*)nullptr, abuf,
                                                   T, C, D, H, W, wd, wh, wwv, sd, sh, sw, 1);
      // qkv GEMM -> bf16
      k_cvt<<<(C * 3 * C + 255) / 256, 256, 0, stream>>>(qkv_w, wbuf, C * 3 * C);
      { dim3 g((3 * C + 63) / 64, (T + 127) / 128);
        k_gemm_bf16<<<g, 256, 0, stream>>>(abuf, wbuf, qkv_b, (float*)nullptr, y16, T, 3 * C, C, 0, 0); }
      // fused attention
      { int QT = (N + 15) / 16;
        k_attn<<<nWin * nh * QT, 32, 0, stream>>>(y16, rpb, aout, nWin, nh, N, C, scale,
                                                  D, H, W, wd, wh, wwv, sd, sh, sw); }
      // proj GEMM -> f32 (window order), then scatter-add into residual stream
      k_cvt<<<(C * C + 255) / 256, 256, 0, stream>>>(prj_w, wbuf, C * C);
      { dim3 g((C + 63) / 64, (T + 127) / 128);
        k_gemm_bf16<<<g, 256, 0, stream>>>(aout, wbuf, prj_b, ybuf, (u16*)nullptr, T, C, C, 0, 0); }
      k_scatter_add<<<(int)(((size_t)T * C + 255) / 256), 256, 0, stream>>>(
          xbuf, ybuf, T, C, D, H, W, wd, wh, wwv, sd, sh, sw);
      // MLP
      k_ln_gather<<<(T + 7) / 8, 256, 0, stream>>>(xbuf, n2g, n2b, (float*)nullptr, abuf,
                                                   T, C, 0, 0, 0, 0, 0, 0, 0, 0, 0, 0);
      k_cvt<<<(C * 4 * C + 255) / 256, 256, 0, stream>>>(fc1_w, wbuf, C * 4 * C);
      { dim3 g((4 * C + 63) / 64, (T + 127) / 128);
        k_gemm_bf16<<<g, 256, 0, stream>>>(abuf, wbuf, fc1_b, (float*)nullptr, y16, T, 4 * C, C, 1, 0); }
      k_cvt<<<(4 * C * C + 255) / 256, 256, 0, stream>>>(fc2_w, wbuf, 4 * C * C);
      { dim3 g((C + 63) / 64, (T + 127) / 128);
        k_gemm_bf16<<<g, 256, 0, stream>>>(y16, wbuf, fc2_b, xbuf, (u16*)nullptr, T, C, 4 * C, 0, 1); }
    }
    if (i < 3) {
      const float* mg  = (const float*)d_in[p + 0];
      const float* mb  = (const float*)d_in[p + 1];
      const float* mrw = (const float*)d_in[p + 2];
      p += 3;
      int H2 = H / 2, W2 = W / 2, Tm = D * H2 * W2, C4 = 4 * C, C2 = 2 * C;
      { size_t tot = (size_t)Tm * C4;
        k_merge_gather<<<(int)((tot + 255) / 256), 256, 0, stream>>>(xbuf, ybuf, D, H, W, C); }
      k_ln_gather<<<(Tm + 7) / 8, 256, 0, stream>>>(ybuf, mg, mb, (float*)nullptr, abuf,
                                                    Tm, C4, 0, 0, 0, 0, 0, 0, 0, 0, 0, 0);
      k_cvt<<<(C4 * C2 + 255) / 256, 256, 0, stream>>>(mrw, wbuf, C4 * C2);
      { dim3 g((C2 + 63) / 64, (Tm + 127) / 128);
        k_gemm_bf16<<<g, 256, 0, stream>>>(abuf, wbuf, (const float*)nullptr, xbuf, (u16*)nullptr,
                                           Tm, C2, C4, 0, 0); }
      H = H2; W = W2; C = 2 * C;
    }
  }
  // final activations: (1, 8, 7, 7, 768) f32
  hipMemcpyAsync(d_out, xbuf, (size_t)D * H * W * C * sizeof(float),
                 hipMemcpyDeviceToDevice, stream);
}
